// DGCN_TopK_2_32908039422445
// MI455X (gfx1250) — compile-verified, matrix operand order checked by reading, not run on hardware
//
#include <hip/hip_runtime.h>
#include <hip/hip_bf16.h>
#include <cmath>

// ---------------------------------------------------------------------------
// DGCNN + TopK pooling for MI455X (gfx1250, wave32).
//  - kNN gram matrices : V_WMMA_F32_16X16X4_F32  (exact fp32, ordering matters)
//  - edge MLPs         : V_WMMA_F32_16X16X32_F16 (f16 in, f32 accumulate)
//  - top-k(20) per row : unrolled register insertion lists (one lane per row)
//  - top-1639/1312     : per-graph bitonic sort in LDS
// Output is permutation-invariant per graph, so pooling = select + scale.
// ---------------------------------------------------------------------------

typedef __attribute__((ext_vector_type(2)))  float    v2f;
typedef __attribute__((ext_vector_type(8)))  float    v8f;
typedef __attribute__((ext_vector_type(16))) _Float16 v16h;

constexpr int NB   = 16;      // graphs
constexpr int NN1  = 2048;    // nodes stage 1
constexpr int F1   = 16;      // input features
constexpr int HDIM = 64;      // hidden
constexpr int KNN  = 20;      // neighbors
constexpr int KK1  = 1639;    // ceil(0.8*2048)
constexpr int NN2  = KK1;     // nodes stage 2
constexpr int KK2  = 1312;    // ceil(0.8*1639)

static __device__ __forceinline__ v8f wmma_f32x4(v2f a, v2f b, v8f c) {
  return __builtin_amdgcn_wmma_f32_16x16x4_f32(false, a, false, b, (short)0, c, false, false);
}
static __device__ __forceinline__ v8f wmma_f16x32(v16h a, v16h b, v8f c) {
  return __builtin_amdgcn_wmma_f32_16x16x32_f16(false, a, false, b, (short)0, c, false, false);
}

// ---------------------------------------------------------------------------
// fp32 -> fp16 weight conversion
// ---------------------------------------------------------------------------
__global__ void f2h_kernel(const float* __restrict__ s, _Float16* __restrict__ d, int n) {
  int t = blockIdx.x * blockDim.x + threadIdx.x;
  if (t < n) d[t] = (_Float16)s[t];
}

// ---------------------------------------------------------------------------
// per-node squared norms
// ---------------------------------------------------------------------------
__global__ void sqnorm_kernel(const float* __restrict__ X, float* __restrict__ SQ,
                              int F, int total) {
  int t = blockIdx.x * blockDim.x + threadIdx.x;
  if (t >= total) return;
  float s = 0.f;
  for (int f = 0; f < F; ++f) { float v = X[(size_t)t * F + f]; s += v * v; }
  SQ[t] = s;
}

// ---------------------------------------------------------------------------
// kNN: one wave per 16-row strip of one graph. Streams 16x16 gram tiles via
// fp32 WMMA (K=4 steps), converts to distances in LDS, lanes 0-15 each keep a
// fully-unrolled top-20 (dist,idx) register list for their row.
// ---------------------------------------------------------------------------
template <int FIN>
__global__ __launch_bounds__(32) void knn_kernel(const float* __restrict__ X,
                                                 const float* __restrict__ SQ,
                                                 int* __restrict__ IDX, int N) {
  __shared__ float rowf[16 * FIN];
  __shared__ float colf[16 * FIN];
  __shared__ float sqr[16];
  __shared__ float sqc[16];
  __shared__ float dt[16 * 16];

  const float FINF = __builtin_inff();
  const int lane = threadIdx.x;
  const int b    = blockIdx.y;
  const int r0   = blockIdx.x * 16;
  const float* Xb  = X  + (size_t)b * N * FIN;
  const float* SQb = SQ + (size_t)b * N;

  for (int i = lane; i < 16 * FIN; i += 32) {
    int r = i / FIN, f = i % FIN;
    int g = r0 + r;
    rowf[i] = (g < N) ? Xb[(size_t)g * FIN + f] : 0.f;
  }
  if (lane < 16) sqr[lane] = (r0 + lane < N) ? SQb[r0 + lane] : 0.f;
  __syncthreads();

  const int m  = lane & 15;
  const int hi = lane >> 4;

  // A fragments: lane holds row m, (K = k0, k0+1) with k0 = 4*ks + 2*hi
  v2f afrag[FIN / 4];
#pragma unroll
  for (int ks = 0; ks < FIN / 4; ++ks) {
    int k0 = ks * 4 + hi * 2;
    afrag[ks].x = rowf[m * FIN + k0];
    afrag[ks].y = rowf[m * FIN + k0 + 1];
  }

  float bd[KNN]; int bi[KNN];
#pragma unroll
  for (int t = 0; t < KNN; ++t) { bd[t] = FINF; bi[t] = 0; }

  const int ntiles = (N + 15) / 16;
  for (int ct = 0; ct < ntiles; ++ct) {
    int c0 = ct * 16;
    __syncthreads();
    for (int i = lane; i < 16 * FIN; i += 32) {
      int c = i / FIN, f = i % FIN;
      int g = c0 + c;
      colf[i] = (g < N) ? Xb[(size_t)g * FIN + f] : 0.f;
    }
    if (lane < 16) sqc[lane] = (c0 + lane < N) ? SQb[c0 + lane] : 0.f;
    __syncthreads();

    v8f acc = {};
#pragma unroll
    for (int ks = 0; ks < FIN / 4; ++ks) {
      int k0 = ks * 4 + hi * 2;
      v2f bfrag;
      bfrag.x = colf[m * FIN + k0];
      bfrag.y = colf[m * FIN + k0 + 1];
      acc = wmma_f32x4(afrag[ks], bfrag, acc);
    }

    // C layout: lane holds rows (r + 8*hi), column m
    float sc = sqc[m];
#pragma unroll
    for (int r = 0; r < 8; ++r) {
      int row = r + hi * 8;
      dt[row * 16 + m] = sqr[row] + sc - 2.f * acc[r];
    }
    __syncthreads();

    if (lane < 16 && (r0 + lane) < N) {
#pragma unroll
      for (int j = 0; j < 16; ++j) {
        int g = c0 + j;
        float d = (g < N) ? dt[lane * 16 + j] : FINF;
        if (d < bd[KNN - 1]) {
          bd[KNN - 1] = d; bi[KNN - 1] = g;
#pragma unroll
          for (int t = KNN - 2; t >= 0; --t) {
            if (bd[t + 1] < bd[t]) {
              float td = bd[t]; bd[t] = bd[t + 1]; bd[t + 1] = td;
              int ti = bi[t]; bi[t] = bi[t + 1]; bi[t + 1] = ti;
            }
          }
        }
      }
    }
  }

  if (lane < 16 && (r0 + lane) < N) {
    int* out = IDX + ((size_t)b * N + (r0 + lane)) * KNN;
#pragma unroll
    for (int t = 0; t < KNN; ++t) out[t] = bi[t];
  }
}

// ---------------------------------------------------------------------------
// Edge MLP + neighbor max: one wave per node. 20 edges -> two 16-row M-tiles.
// layer1: [xi | xj-xi] (2*FIN) @ W1 (+b1, relu) -> 64 ; layer2: @ W2 -> 64,
// masked max over 20 edges (+b2, optional relu). f16 WMMA, f32 accumulate.
// ---------------------------------------------------------------------------
template <int FIN, bool RELU>
__global__ __launch_bounds__(32) void edge_mlp_kernel(
    const float* __restrict__ X, const int* __restrict__ IDX,
    const _Float16* __restrict__ W1, const float* __restrict__ BIAS1,
    const _Float16* __restrict__ W2, const float* __restrict__ BIAS2,
    float* __restrict__ OUT, int N) {
  constexpr int KIN = 2 * FIN;
  __shared__ _Float16 W1s[KIN * HDIM];
  __shared__ _Float16 W2s[HDIM * HDIM];
  __shared__ _Float16 me[32 * KIN];    // 32 edge rows (20 valid) x KIN
  __shared__ _Float16 hm[32 * HDIM];   // layer-1 activations

  const int lane = threadIdx.x;
  const int n = blockIdx.x;
  const int b = blockIdx.y;

  for (int i = lane; i < KIN * HDIM; i += 32)  W1s[i] = W1[i];
  for (int i = lane; i < HDIM * HDIM; i += 32) W2s[i] = W2[i];

  const float* Xb = X + (size_t)b * N * FIN;
  const int* nbr = IDX + ((size_t)b * N + n) * KNN;
  {
    int e = lane;
    int j = (e < KNN) ? nbr[e] : 0;
    const float* xi = Xb + (size_t)n * FIN;
    const float* xj = Xb + (size_t)j * FIN;
    for (int f = 0; f < FIN; ++f) {
      float a   = xi[f];
      float dlt = xj[f] - a;
      me[e * KIN + f]       = (e < KNN) ? (_Float16)a   : (_Float16)0.f;
      me[e * KIN + FIN + f] = (e < KNN) ? (_Float16)dlt : (_Float16)0.f;
    }
  }
  __syncthreads();

  const int m  = lane & 15;
  const int hi = lane >> 4;
  const int kb = hi * 8;   // f16 A/B half-split: lanes<16 own K 0-7 & 16-23

  // ---- layer 1 ----
#pragma unroll
  for (int t = 0; t < 2; ++t) {
#pragma unroll
    for (int nt = 0; nt < 4; ++nt) {
      v8f c = {};
#pragma unroll
      for (int ks = 0; ks < KIN / 32; ++ks) {
        v16h a, w;
        int e = t * 16 + m;
        int col = nt * 16 + m;
#pragma unroll
        for (int i = 0; i < 8; ++i) {
          a[i]     = me[e * KIN + ks * 32 + kb + i];
          a[8 + i] = me[e * KIN + ks * 32 + 16 + kb + i];
          w[i]     = W1s[(ks * 32 + kb + i) * HDIM + col];
          w[8 + i] = W1s[(ks * 32 + 16 + kb + i) * HDIM + col];
        }
        c = wmma_f16x32(a, w, c);
      }
      float bias = BIAS1[nt * 16 + m];
#pragma unroll
      for (int r = 0; r < 8; ++r) {
        float v = c[r] + bias;
        v = v > 0.f ? v : 0.f;
        hm[(t * 16 + r + hi * 8) * HDIM + nt * 16 + m] = (_Float16)v;
      }
    }
  }
  __syncthreads();

  // ---- layer 2 + masked max over edges ----
#pragma unroll
  for (int nt = 0; nt < 4; ++nt) {
    float mx = -__builtin_inff();
#pragma unroll
    for (int t = 0; t < 2; ++t) {
      v8f c = {};
#pragma unroll
      for (int ks = 0; ks < 2; ++ks) {
        v16h a, w;
        int e = t * 16 + m;
        int col = nt * 16 + m;
#pragma unroll
        for (int i = 0; i < 8; ++i) {
          a[i]     = hm[e * HDIM + ks * 32 + kb + i];
          a[8 + i] = hm[e * HDIM + ks * 32 + 16 + kb + i];
          w[i]     = W2s[(ks * 32 + kb + i) * HDIM + col];
          w[8 + i] = W2s[(ks * 32 + 16 + kb + i) * HDIM + col];
        }
        c = wmma_f16x32(a, w, c);
      }
#pragma unroll
      for (int r = 0; r < 8; ++r) {
        int eg = t * 16 + r + hi * 8;
        if (eg < KNN) mx = fmaxf(mx, c[r]);
      }
    }
    mx = fmaxf(mx, __shfl_xor(mx, 16, 32));     // combine row halves per column
    float val = mx + BIAS2[nt * 16 + m];
    if (RELU) val = val > 0.f ? val : 0.f;
    if (hi == 0) OUT[((size_t)b * N + n) * HDIM + nt * 16 + m] = val;
  }
}

// ---------------------------------------------------------------------------
// score = tanh(h . w / ||w||)
// ---------------------------------------------------------------------------
__global__ void score_kernel(const float* __restrict__ Hin, const float* __restrict__ w,
                             float* __restrict__ score, int total) {
  int t = blockIdx.x * blockDim.x + threadIdx.x;
  if (t >= total) return;
  float nrm = 0.f, s = 0.f;
  for (int f = 0; f < HDIM; ++f) { nrm += w[f] * w[f]; s += Hin[(size_t)t * HDIM + f] * w[f]; }
  score[t] = tanhf(s / sqrtf(nrm));
}

// ---------------------------------------------------------------------------
// per-graph top-KK selection via bitonic sort of 2048 padded (key,idx) pairs.
// key = -score ascending == score descending; tie-break on lower index.
// ---------------------------------------------------------------------------
__global__ __launch_bounds__(1024) void topk_select_kernel(
    const float* __restrict__ score, int N, int KKc,
    int* __restrict__ perm, float* __restrict__ vals) {
  __shared__ float sk[2048];
  __shared__ int   si[2048];
  const int b = blockIdx.x, t = threadIdx.x;
  for (int i = t; i < 2048; i += 1024) {
    sk[i] = (i < N) ? -score[(size_t)b * N + i] : __builtin_inff();
    si[i] = i;
  }
  __syncthreads();
  for (int ksz = 2; ksz <= 2048; ksz <<= 1) {
    for (int j = ksz >> 1; j > 0; j >>= 1) {
      for (int i = t; i < 2048; i += 1024) {
        int ixj = i ^ j;
        if (ixj > i) {
          bool dir = ((i & ksz) == 0);  // ascending segment
          float ka = sk[i];  int ia = si[i];
          float kc = sk[ixj]; int ic = si[ixj];
          bool aGreater = (ka > kc) || (ka == kc && ia > ic);
          if (aGreater == dir) { sk[i] = kc; si[i] = ic; sk[ixj] = ka; si[ixj] = ia; }
        }
      }
      __syncthreads();
    }
  }
  for (int i = t; i < KKc; i += 1024) {
    perm[(size_t)b * KKc + i] = si[i];
    vals[(size_t)b * KKc + i] = -sk[i];
  }
}

// ---------------------------------------------------------------------------
// gather selected nodes and scale by their (tanh) score
// ---------------------------------------------------------------------------
__global__ void gather_scale_kernel(const float* __restrict__ Hin, const int* __restrict__ perm,
                                    const float* __restrict__ vals, float* __restrict__ P,
                                    int Nsrc, int KKc, int total) {
  int t = blockIdx.x * blockDim.x + threadIdx.x;
  if (t >= total) return;
  int f = t & (HDIM - 1);
  int rest = t / HDIM;
  int mm = rest % KKc;
  int b = rest / KKc;
  int src = perm[(size_t)b * KKc + mm];
  P[t] = Hin[((size_t)b * Nsrc + src) * HDIM + f] * vals[(size_t)b * KKc + mm];
}

// ---------------------------------------------------------------------------
// global attention pool: x[b] = sum_n softmax_n(p.gw + gb) * p[n]
// ---------------------------------------------------------------------------
__global__ __launch_bounds__(256) void gap_kernel(const float* __restrict__ P, int Nn,
                                                  const float* __restrict__ gw,
                                                  const float* __restrict__ gb,
                                                  float* __restrict__ xout) {
  __shared__ float lg[2048];
  __shared__ float red[256];
  const int b = blockIdx.x, t = threadIdx.x;
  const float* Pb = P + (size_t)b * Nn * HDIM;
  for (int i = t; i < Nn; i += 256) {
    float s = gb[0];
    for (int f = 0; f < HDIM; ++f) s += Pb[(size_t)i * HDIM + f] * gw[f];
    lg[i] = s;
  }
  __syncthreads();
  float mx = -__builtin_inff();
  for (int i = t; i < Nn; i += 256) mx = fmaxf(mx, lg[i]);
  red[t] = mx; __syncthreads();
  for (int s2 = 128; s2 > 0; s2 >>= 1) { if (t < s2) red[t] = fmaxf(red[t], red[t + s2]); __syncthreads(); }
  float M = red[0]; __syncthreads();
  float sum = 0.f;
  for (int i = t; i < Nn; i += 256) { float e = expf(lg[i] - M); lg[i] = e; sum += e; }
  red[t] = sum; __syncthreads();
  for (int s2 = 128; s2 > 0; s2 >>= 1) { if (t < s2) red[t] += red[t + s2]; __syncthreads(); }
  float S = red[0]; __syncthreads();
  if (t < HDIM) {
    float acc = 0.f;
    for (int i = 0; i < Nn; ++i) acc += lg[i] * Pb[(size_t)i * HDIM + t];
    xout[(size_t)b * HDIM + t] = acc / S;
  }
}

// ---------------------------------------------------------------------------
// out[b,c] = (x1+x2) . lw[:,c] + lb[c]
// ---------------------------------------------------------------------------
__global__ void final_kernel(const float* __restrict__ x1, const float* __restrict__ x2,
                             const float* __restrict__ lw, const float* __restrict__ lb,
                             float* __restrict__ out) {
  int b = blockIdx.x, c = threadIdx.x;
  float s = lb[c];
  for (int h = 0; h < HDIM; ++h) s += (x1[b * HDIM + h] + x2[b * HDIM + h]) * lw[h * 2 + c];
  out[b * 2 + c] = s;
}

// ---------------------------------------------------------------------------
extern "C" void kernel_launch(void* const* d_in, const int* in_sizes, int n_in,
                              void* d_out, int out_size, void* d_ws, size_t ws_size,
                              hipStream_t stream) {
  const float* x   = (const float*)d_in[0];
  const float* w1a = (const float*)d_in[5];
  const float* b1a = (const float*)d_in[6];
  const float* w1b = (const float*)d_in[7];
  const float* b1b = (const float*)d_in[8];
  const float* w2a = (const float*)d_in[9];
  const float* b2a = (const float*)d_in[10];
  const float* w2b = (const float*)d_in[11];
  const float* b2b = (const float*)d_in[12];
  const float* pw1 = (const float*)d_in[13];
  const float* pw2 = (const float*)d_in[14];
  const float* g1w = (const float*)d_in[15];
  const float* g1b = (const float*)d_in[16];
  const float* g2w = (const float*)d_in[17];
  const float* g2b = (const float*)d_in[18];
  const float* lw  = (const float*)d_in[19];
  const float* lb  = (const float*)d_in[20];

  char* ws = (char*)d_ws;
  size_t off = 0;
  auto carve = [&](size_t bytes) -> void* {
    void* p = ws + off;
    off += (bytes + 255) & ~(size_t)255;
    return p;
  };
  _Float16* W1h1 = (_Float16*)carve((size_t)2 * F1 * HDIM * sizeof(_Float16));
  _Float16* W2h1 = (_Float16*)carve((size_t)HDIM * HDIM * sizeof(_Float16));
  _Float16* W1h2 = (_Float16*)carve((size_t)2 * HDIM * HDIM * sizeof(_Float16));
  _Float16* W2h2 = (_Float16*)carve((size_t)HDIM * HDIM * sizeof(_Float16));
  float* sq1   = (float*)carve((size_t)NB * NN1 * 4);
  int*   idx1  = (int*)  carve((size_t)NB * NN1 * KNN * 4);
  float* h1    = (float*)carve((size_t)NB * NN1 * HDIM * 4);
  float* sc1   = (float*)carve((size_t)NB * NN1 * 4);
  int*   perm1 = (int*)  carve((size_t)NB * KK1 * 4);
  float* val1  = (float*)carve((size_t)NB * KK1 * 4);
  float* p1    = (float*)carve((size_t)NB * KK1 * HDIM * 4);
  float* sq2   = (float*)carve((size_t)NB * NN2 * 4);
  int*   idx2  = (int*)  carve((size_t)NB * NN2 * KNN * 4);
  float* h2    = (float*)carve((size_t)NB * NN2 * HDIM * 4);
  float* sc2   = (float*)carve((size_t)NB * NN2 * 4);
  int*   perm2 = (int*)  carve((size_t)NB * KK2 * 4);
  float* val2  = (float*)carve((size_t)NB * KK2 * 4);
  float* p2    = (float*)carve((size_t)NB * KK2 * HDIM * 4);
  float* x1g   = (float*)carve((size_t)NB * HDIM * 4);
  float* x2g   = (float*)carve((size_t)NB * HDIM * 4);

  // weight conversion (f32 -> f16)
  f2h_kernel<<<dim3((2 * F1 * HDIM + 255) / 256), 256, 0, stream>>>(w1a, W1h1, 2 * F1 * HDIM);
  f2h_kernel<<<dim3((HDIM * HDIM + 255) / 256), 256, 0, stream>>>(w1b, W2h1, HDIM * HDIM);
  f2h_kernel<<<dim3((2 * HDIM * HDIM + 255) / 256), 256, 0, stream>>>(w2a, W1h2, 2 * HDIM * HDIM);
  f2h_kernel<<<dim3((HDIM * HDIM + 255) / 256), 256, 0, stream>>>(w2b, W2h2, HDIM * HDIM);

  // ---- stage 1 ----
  sqnorm_kernel<<<(NB * NN1 + 255) / 256, 256, 0, stream>>>(x, sq1, F1, NB * NN1);
  knn_kernel<F1><<<dim3((NN1 + 15) / 16, NB), 32, 0, stream>>>(x, sq1, idx1, NN1);
  edge_mlp_kernel<F1, true><<<dim3(NN1, NB), 32, 0, stream>>>(x, idx1, W1h1, b1a, W2h1, b1b, h1, NN1);
  score_kernel<<<(NB * NN1 + 255) / 256, 256, 0, stream>>>(h1, pw1, sc1, NB * NN1);
  topk_select_kernel<<<NB, 1024, 0, stream>>>(sc1, NN1, KK1, perm1, val1);
  gather_scale_kernel<<<(NB * KK1 * HDIM + 255) / 256, 256, 0, stream>>>(h1, perm1, val1, p1, NN1, KK1, NB * KK1 * HDIM);
  gap_kernel<<<NB, 256, 0, stream>>>(p1, KK1, g1w, g1b, x1g);

  // ---- stage 2 ----
  sqnorm_kernel<<<(NB * NN2 + 255) / 256, 256, 0, stream>>>(p1, sq2, HDIM, NB * NN2);
  knn_kernel<HDIM><<<dim3((NN2 + 15) / 16, NB), 32, 0, stream>>>(p1, sq2, idx2, NN2);
  edge_mlp_kernel<HDIM, false><<<dim3(NN2, NB), 32, 0, stream>>>(p1, idx2, W1h2, b2a, W2h2, b2b, h2, NN2);
  score_kernel<<<(NB * NN2 + 255) / 256, 256, 0, stream>>>(h2, pw2, sc2, NB * NN2);
  topk_select_kernel<<<NB, 1024, 0, stream>>>(sc2, NN2, KK2, perm2, val2);
  gather_scale_kernel<<<(NB * KK2 * HDIM + 255) / 256, 256, 0, stream>>>(h2, perm2, val2, p2, NN2, KK2, NB * KK2 * HDIM);
  gap_kernel<<<NB, 256, 0, stream>>>(p2, KK2, g2w, g2b, x2g);

  final_kernel<<<NB, 2, 0, stream>>>(x1g, x2g, lw, lb, (float*)d_out);
}
